// NeuralCDE_43705587204612
// MI455X (gfx1250) — matrix operand order repaired and checked by reading, not compiled, not run on hardware
//
#include <hip/hip_runtime.h>
#include <hip/hip_bf16.h>

typedef __attribute__((ext_vector_type(16))) _Float16 v16h;
typedef __attribute__((ext_vector_type(8)))  _Float16 v8h;
typedef __attribute__((ext_vector_type(8)))  float    v8f;

#define BB 1024
#define TT 64
#define DD 8
#define HH 64
#define WW 128
#define BM 16
#define NTHREADS 256

// Tsit5 tableau
#define A21f 0.161f
#define A31f (-0.008480655492356989f)
#define A32f 0.335480655492357f
#define A41f 2.8971530571054935f
#define A42f (-6.359448489975075f)
#define A43f 4.3622954328695815f
#define A51f 5.325864828439257f
#define A52f (-11.748883564062828f)
#define A53f 7.4955393428898365f
#define A54f (-0.09249506636175525f)
#define A61f 5.86145544294642f
#define A62f (-12.92096931784711f)
#define A63f 8.159367898576159f
#define A64f (-0.071584973281401f)
#define A65f (-0.028269050394068383f)
#define B1f 0.09646076681806523f
#define B2f 0.01f
#define B3f 0.4798896504144996f
#define B4f 1.379008574103742f
#define B5f (-3.290069515436081f)
#define B6f 2.324710524099774f

// zero-padded per-stage coefficients: stage s uses rows SC[s][0..4]
__constant__ const float SC[6][5] = {
    {0.f, 0.f, 0.f, 0.f, 0.f},
    {A21f, 0.f, 0.f, 0.f, 0.f},
    {A31f, A32f, 0.f, 0.f, 0.f},
    {A41f, A42f, A43f, 0.f, 0.f},
    {A51f, A52f, A53f, A54f, 0.f},
    {A61f, A62f, A63f, A64f, A65f}};

// LDS layout (bytes)
#define OFF_W1H  0        // 128*64  f16 = 16384
#define OFF_W2H  16384    // 128*128 f16 = 32768
#define OFF_ACT0 49152    // 16*64   f16 = 2048
#define OFF_ACT1 51200    // 16*128  f16 = 4096
#define OFF_ACT2 55296    // 16*128  f16 = 4096
#define OFF_B1   59392    // 128 f32
#define OFF_B2   59904    // 128 f32
#define OFF_B3   60416    // 512 f32
#define OFF_K    62464    // 6*1024 f32 = 24576
#define OFF_Y    87040    // 1024 f32
#define OFF_DX   91136    // 128 f32
#define OFF_TS   91648    // 64 f32
#define SMEM_BYTES 91904

#define LOG2E 1.44269504f
#define LN2   0.69314718f

// ---- bare-metal transcendentals: raw v_exp_f32 / v_log_f32 / v_rcp_f32,
// no libm range-reduction or denorm guards (arguments provably safe).
__device__ __forceinline__ float softplus_f(float x) {
    // max(x,0) + ln2 * log2(1 + 2^(-log2e*|x|));  log arg in (1,2] -> never denorm
    float t = __builtin_amdgcn_exp2f(-LOG2E * fabsf(x));
    float l = __builtin_amdgcn_logf(1.0f + t);
    return fmaf(LN2, l, fmaxf(x, 0.0f));
}
__device__ __forceinline__ float tanh_f(float x) {
    // tanh = 1 - 2/(exp(2x)+1); clamp keeps exp finite, saturates to +/-1
    float xc = fminf(fmaxf(x, -10.0f), 10.0f);
    float e = __builtin_amdgcn_exp2f((2.0f * LOG2E) * xc);
    return fmaf(-2.0f, __builtin_amdgcn_rcpf(e + 1.0f), 1.0f);
}
__device__ __forceinline__ float sigmoid_f(float x) {
    return __builtin_amdgcn_rcpf(1.0f + __builtin_amdgcn_exp2f(-LOG2E * x));
}

// A-operand fragment: lane holds row m = lane&15; contiguous K order,
// k = 16*(lane-half) + s, matching B. Two aligned 16B LDS loads.
__device__ __forceinline__ v16h load_a_frag(const _Float16* base, int pitch,
                                            int m, int kb, int hf) {
    v8h lo = *(const v8h*)(base + m * pitch + kb + 16 * hf);
    v8h hi = *(const v8h*)(base + m * pitch + kb + 16 * hf + 8);
    return __builtin_shufflevector(lo, hi, 0, 1, 2, 3, 4, 5, 6, 7,
                                   8, 9, 10, 11, 12, 13, 14, 15);
}

// One vector-field evaluation for a 16-row batch tile.
// act0 [16x64] f16 in  -> kout [16x64] f32 (einsum with dxb already applied)
__device__ void vf_eval(const _Float16* act0, _Float16* act1, _Float16* act2,
                        const _Float16* w1h, const _Float16* w2h,
                        const _Float16* __restrict__ w3h,
                        const float* b1s, const float* b2s, const float* b3s,
                        const float* dxb, float* kout, int lane, int wv) {
    const int hf = lane >> 4;
    const int n16 = lane & 15;

    // ---- layer 1: [16x64] @ [64x128]^T, softplus -> act1
    {
        const int nb = wv * 16;
        v8f acc = {0.f, 0.f, 0.f, 0.f, 0.f, 0.f, 0.f, 0.f};
#pragma unroll
        for (int kb = 0; kb < HH; kb += 32) {
            v16h a = load_a_frag(act0, HH, n16, kb, hf);
            // B[k][n] = w1[n][k]: row-major weight row IS the B column -> 32B load
            v16h b = *(const v16h*)(w1h + (nb + n16) * HH + kb + 16 * hf);
            acc = __builtin_amdgcn_wmma_f32_16x16x32_f16(false, a, false, b,
                                                         (short)0, acc, false, false);
        }
        const float bias = b1s[nb + n16];
#pragma unroll
        for (int r = 0; r < 8; ++r)
            act1[(r + 8 * hf) * WW + nb + n16] = (_Float16)softplus_f(acc[r] + bias);
    }
    __syncthreads();

    // ---- layer 2: [16x128] @ [128x128]^T, softplus -> act2
    {
        const int nb = wv * 16;
        v8f acc = {0.f, 0.f, 0.f, 0.f, 0.f, 0.f, 0.f, 0.f};
#pragma unroll
        for (int kb = 0; kb < WW; kb += 32) {
            v16h a = load_a_frag(act1, WW, n16, kb, hf);
            v16h b = *(const v16h*)(w2h + (nb + n16) * WW + kb + 16 * hf);
            acc = __builtin_amdgcn_wmma_f32_16x16x32_f16(false, a, false, b,
                                                         (short)0, acc, false, false);
        }
        const float bias = b2s[nb + n16];
#pragma unroll
        for (int r = 0; r < 8; ++r)
            act2[(r + 8 * hf) * WW + nb + n16] = (_Float16)softplus_f(acc[r] + bias);
    }
    __syncthreads();

    // ---- layer 3: [16x128] @ [128x512]^T, tanh, contract over D with dx
    // w3 has zero intra-eval reuse per byte -> stream straight from L2 (hot),
    // per-lane contiguous 32B global_load_b128 pairs.
#pragma unroll
    for (int tt = 0; tt < 4; ++tt) {
        const int nb = (wv * 4 + tt) * 16;
        v8f acc = {0.f, 0.f, 0.f, 0.f, 0.f, 0.f, 0.f, 0.f};
#pragma unroll
        for (int kb = 0; kb < WW; kb += 32) {
            v16h a = load_a_frag(act2, WW, n16, kb, hf);
            v16h b = *(const v16h*)(w3h + (size_t)(nb + n16) * WW + kb + 16 * hf);
            acc = __builtin_amdgcn_wmma_f32_16x16x32_f16(false, a, false, b,
                                                         (short)0, acc, false, false);
        }
        const int nn = nb + n16;
        const float bias = b3s[nn];
        const int ho = nn >> 3;   // hidden index
        const int di = nn & 7;    // data-channel index
#pragma unroll
        for (int r = 0; r < 8; ++r) {
            const int m = r + 8 * hf;
            float v = tanh_f(acc[r] + bias) * dxb[m * DD + di];
            // reduce over D=8 (lanes grouped by 8 share (m,ho))
            v += __shfl_xor(v, 1);
            v += __shfl_xor(v, 2);
            v += __shfl_xor(v, 4);
            if ((lane & 7) == 0) kout[m * HH + ho] = v;
        }
    }
}

__global__ void __launch_bounds__(NTHREADS) cde_main(
    const float* __restrict__ ts, const float* __restrict__ xs,
    const float* __restrict__ iw1, const float* __restrict__ ib1,
    const float* __restrict__ iw2, const float* __restrict__ ib2,
    const float* __restrict__ iw3, const float* __restrict__ ib3,
    const float* __restrict__ vw1, const float* __restrict__ vb1,
    const float* __restrict__ vw2, const float* __restrict__ vb2,
    const float* __restrict__ vb3, const _Float16* __restrict__ w3h,
    const float* __restrict__ lw, const float* __restrict__ lb,
    float* __restrict__ out) {
    extern __shared__ char smem[];
    _Float16* w1h  = (_Float16*)(smem + OFF_W1H);
    _Float16* w2h  = (_Float16*)(smem + OFF_W2H);
    _Float16* act0 = (_Float16*)(smem + OFF_ACT0);
    _Float16* act1 = (_Float16*)(smem + OFF_ACT1);
    _Float16* act2 = (_Float16*)(smem + OFF_ACT2);
    float* b1s  = (float*)(smem + OFF_B1);
    float* b2s  = (float*)(smem + OFF_B2);
    float* b3s  = (float*)(smem + OFF_B3);
    float* kbuf = (float*)(smem + OFF_K);
    float* ybuf = (float*)(smem + OFF_Y);
    float* dxb  = (float*)(smem + OFF_DX);
    float* tsl  = (float*)(smem + OFF_TS);

    const int tid = threadIdx.x;
    const int lane = tid & 31;
    const int wv = tid >> 5;
    const int b0 = blockIdx.x * BM;

    // Pre-warm w3h into the cache hierarchy (global_prefetch_b8) so the first
    // L3 GEMM doesn't cold-miss: 1024 cachelines of 128B across 256 threads.
    for (int i = tid * 64; i < HH * DD * WW; i += NTHREADS * 64)
        __builtin_prefetch(w3h + i, 0, 0);

    // ---- stage weights/biases into LDS (f16 weights)
    for (int i = tid; i < WW * HH; i += NTHREADS) w1h[i] = (_Float16)vw1[i];
    for (int i = tid; i < WW * WW; i += NTHREADS) w2h[i] = (_Float16)vw2[i];
    for (int i = tid; i < WW; i += NTHREADS) { b1s[i] = vb1[i]; b2s[i] = vb2[i]; }
    for (int i = tid; i < HH * DD; i += NTHREADS) b3s[i] = vb3[i];
    for (int i = tid; i < TT; i += NTHREADS) tsl[i] = ts[i];
    __syncthreads();

    // ---- initial MLP (VALU, one-time): D(8) -> 128 relu -> 128 relu -> 64
    float* t1 = kbuf;          // reuse kbuf as scratch: 16x128
    float* t2 = kbuf + 2048;   // 16x128
    for (int idx = tid; idx < BM * WW; idx += NTHREADS) {
        const int m = idx >> 7, n = idx & 127;
        const float* xr = xs + (size_t)(b0 + m) * TT * DD;  // t = 0
        float s = ib1[n];
#pragma unroll
        for (int k = 0; k < DD; ++k) s += xr[k] * iw1[n * DD + k];
        t1[idx] = fmaxf(s, 0.0f);
    }
    __syncthreads();
    for (int idx = tid; idx < BM * WW; idx += NTHREADS) {
        const int m = idx >> 7, n = idx & 127;
        float s = ib2[n];
        for (int k = 0; k < WW; ++k) s += t1[m * WW + k] * iw2[n * WW + k];
        t2[idx] = fmaxf(s, 0.0f);
    }
    __syncthreads();
    for (int idx = tid; idx < BM * HH; idx += NTHREADS) {
        const int m = idx >> 6, h = idx & 63;
        float s = ib3[h];
        for (int k = 0; k < WW; ++k) s += t2[m * WW + k] * iw3[h * WW + k];
        ybuf[idx] = s;
    }
    __syncthreads();

    // readout at t = 0
    if (tid < BM) {
        float s = lb[0];
        for (int h = 0; h < HH; ++h) s += ybuf[tid * HH + h] * lw[h];
        out[(size_t)(b0 + tid) * TT + 0] = sigmoid_f(s);
    }
    __syncthreads();

    // ---- time stepping (sequential): Tsit5, 6 stages per step
    for (int t = 0; t < TT - 1; ++t) {
        const float dtv = tsl[t + 1] - tsl[t];
        if (tid < BM * DD) {
            const int m = tid >> 3, d = tid & 7;
            const float* xr = xs + ((size_t)(b0 + m) * TT + t) * DD + d;
            dxb[tid] = (xr[DD] - xr[0]) / dtv;
        }
        __syncthreads();

        // rolled stage loop: branch-free zero-padded tableau accumulation
        for (int s = 0; s < 6; ++s) {
            const float c0 = dtv * SC[s][0], c1 = dtv * SC[s][1];
            const float c2 = dtv * SC[s][2], c3 = dtv * SC[s][3];
            const float c4 = dtv * SC[s][4];
            for (int idx = tid; idx < BM * HH; idx += NTHREADS) {
                float v = ybuf[idx];
                v = fmaf(c0, kbuf[0 * BM * HH + idx], v);
                v = fmaf(c1, kbuf[1 * BM * HH + idx], v);
                v = fmaf(c2, kbuf[2 * BM * HH + idx], v);
                v = fmaf(c3, kbuf[3 * BM * HH + idx], v);
                v = fmaf(c4, kbuf[4 * BM * HH + idx], v);
                act0[idx] = (_Float16)v;
            }
            __syncthreads();
            vf_eval(act0, act1, act2, w1h, w2h, w3h, b1s, b2s, b3s, dxb,
                    kbuf + s * (BM * HH), lane, wv);
            __syncthreads();
        }

        // y_{t+1} = y + dt * (B1 k1 + ... + B6 k6)
        for (int idx = tid; idx < BM * HH; idx += NTHREADS) {
            float v = ybuf[idx];
            v = fmaf(dtv * B1f, kbuf[0 * BM * HH + idx], v);
            v = fmaf(dtv * B2f, kbuf[1 * BM * HH + idx], v);
            v = fmaf(dtv * B3f, kbuf[2 * BM * HH + idx], v);
            v = fmaf(dtv * B4f, kbuf[3 * BM * HH + idx], v);
            v = fmaf(dtv * B5f, kbuf[4 * BM * HH + idx], v);
            v = fmaf(dtv * B6f, kbuf[5 * BM * HH + idx], v);
            ybuf[idx] = v;
        }
        __syncthreads();

        if (tid < BM) {
            float s = lb[0];
            for (int h = 0; h < HH; ++h) s += ybuf[tid * HH + h] * lw[h];
            out[(size_t)(b0 + tid) * TT + (t + 1)] = sigmoid_f(s);
        }
        __syncthreads();
    }
}

__global__ void cvt_f16(const float* __restrict__ src, _Float16* __restrict__ dst,
                        int n) {
    const int i = blockIdx.x * blockDim.x + threadIdx.x;
    if (i < n) dst[i] = (_Float16)src[i];
}

extern "C" void kernel_launch(void* const* d_in, const int* in_sizes, int n_in,
                              void* d_out, int out_size, void* d_ws, size_t ws_size,
                              hipStream_t stream) {
    const float* ts  = (const float*)d_in[0];
    const float* xs  = (const float*)d_in[1];
    const float* iw1 = (const float*)d_in[2];
    const float* ib1 = (const float*)d_in[3];
    const float* iw2 = (const float*)d_in[4];
    const float* ib2 = (const float*)d_in[5];
    const float* iw3 = (const float*)d_in[6];
    const float* ib3 = (const float*)d_in[7];
    const float* vw1 = (const float*)d_in[8];
    const float* vb1 = (const float*)d_in[9];
    const float* vw2 = (const float*)d_in[10];
    const float* vb2 = (const float*)d_in[11];
    const float* vw3 = (const float*)d_in[12];
    const float* vb3 = (const float*)d_in[13];
    const float* lw  = (const float*)d_in[14];
    const float* lb  = (const float*)d_in[15];
    float* out = (float*)d_out;

    // vw3 -> f16 copy in workspace (streamed from L2 by the main kernel)
    _Float16* w3h = (_Float16*)d_ws;
    const int n3 = HH * DD * WW;  // 65536
    cvt_f16<<<(n3 + 255) / 256, 256, 0, stream>>>(vw3, w3h, n3);

    cde_main<<<dim3(BB / BM), dim3(NTHREADS), SMEM_BYTES, stream>>>(
        ts, xs, iw1, ib1, iw2, ib2, iw3, ib3, vw1, vb1, vw2, vb2, vb3, w3h,
        lw, lb, out);
}
